// LlamaAttention_19146964206021
// MI455X (gfx1250) — matrix operand order repaired and checked by reading, not compiled, not run on hardware
//
#include <hip/hip_runtime.h>
#include <hip/hip_bf16.h>

// CDNA5 (gfx1250) wave32 WMMA implementation of Llama attention.
// All matmuls run through v_wmma_f32_16x16x32_f16 (f16 in, f32 accum).

typedef __attribute__((ext_vector_type(16))) _Float16 v16h;
typedef __attribute__((ext_vector_type(4)))  _Float16 v4h;
typedef __attribute__((ext_vector_type(8)))  float    v8f;

#define NHq 32
#define NKVq 8
#define HDq 128
#define QH  (NHq * HDq)   // 4096
#define KVH (NKVq * HDq)  // 1024

union F16Frag { uint4 u[2]; v16h v; };

static __device__ __forceinline__ v8f wmma_f16(const v16h& a, const v16h& b, const v8f& c) {
  return __builtin_amdgcn_wmma_f32_16x16x32_f16(false, a, false, b, (short)0, c, false, false);
}

// ---------------------------------------------------------------- cast fp32 -> f16 (x4 vectorized)
__global__ void cast_f32_to_f16_v4(const float* __restrict__ src,
                                   _Float16* __restrict__ dst, long long n4) {
  long long i = (long long)blockIdx.x * blockDim.x + threadIdx.x;
  long long stride = (long long)gridDim.x * blockDim.x;
  const float4* s4 = reinterpret_cast<const float4*>(src);
  v4h* d4 = reinterpret_cast<v4h*>(dst);
  for (; i < n4; i += stride) {
    float4 x = s4[i];
    v4h y = {(_Float16)x.x, (_Float16)x.y, (_Float16)x.z, (_Float16)x.w};
    d4[i] = y;
  }
}

// ---------------------------------------------------------------- GEMM: C[M,N] = A[M,K] * W[N,K]^T
// Block = 128 threads (4 waves, 2x2), wave tile 64x64 (4x4 WMMA frags).
// Double-buffered; sched_barrier pins "issue loads of tile k+32, then MMA
// tile k" so the scheduler can't re-serialize into load/wait/wmma pairs.
template <int F32OUT>
__global__ void __launch_bounds__(128)
gemm_xWT(const _Float16* __restrict__ A, const _Float16* __restrict__ W,
         void* __restrict__ Cout, int M, int N, int K) {
  const int lane = threadIdx.x & 31;
  const int wave = threadIdx.x >> 5;
  const int hi = lane >> 4;
  const int lm = lane & 15;
  const int mbase = blockIdx.y * 128 + (wave >> 1) * 64;
  const int nbase = blockIdx.x * 128 + (wave & 1) * 64;

  v8f acc[4][4] = {};
  F16Frag af0[4], bf0[4], af1[4], bf1[4];

  auto load_tile = [&](int kk, F16Frag (&af)[4], F16Frag (&bf)[4]) {
#pragma unroll
    for (int i = 0; i < 4; i++) {
      const _Float16* ap = A + (size_t)(mbase + i * 16 + lm) * K + kk;
      af[i].u[0] = *reinterpret_cast<const uint4*>(ap + 16 * hi);
      af[i].u[1] = *reinterpret_cast<const uint4*>(ap + 16 * hi + 8);
      if (i == 0) __builtin_prefetch(ap + 512, 0, 1);  // global_prefetch_b8
    }
#pragma unroll
    for (int j = 0; j < 4; j++) {
      const _Float16* wp = W + (size_t)(nbase + j * 16 + lm) * K + kk + 16 * hi;
      bf[j].u[0] = *reinterpret_cast<const uint4*>(wp);
      bf[j].u[1] = *reinterpret_cast<const uint4*>(wp + 8);
      if (j == 0) __builtin_prefetch(wp + 512, 0, 1);
    }
  };
  auto mma_tile = [&](F16Frag (&af)[4], F16Frag (&bf)[4]) {
#pragma unroll
    for (int i = 0; i < 4; i++)
#pragma unroll
      for (int j = 0; j < 4; j++)
        acc[i][j] = wmma_f16(af[i].v, bf[j].v, acc[i][j]);
  };

  // K is a multiple of 64 here (K = 4096)
  load_tile(0, af0, bf0);
  int k = 0;
  while (k < K - 64) {
    load_tile(k + 32, af1, bf1);
    __builtin_amdgcn_sched_barrier(0);
    mma_tile(af0, bf0);
    __builtin_amdgcn_sched_barrier(0);
    load_tile(k + 64, af0, bf0);
    __builtin_amdgcn_sched_barrier(0);
    mma_tile(af1, bf1);
    __builtin_amdgcn_sched_barrier(0);
    k += 64;
  }
  load_tile(k + 32, af1, bf1);
  __builtin_amdgcn_sched_barrier(0);
  mma_tile(af0, bf0);
  mma_tile(af1, bf1);

  // C/D layout: VGPR r of lane L holds C[row = r + 8*hi][col = lm]
#pragma unroll
  for (int i = 0; i < 4; i++)
#pragma unroll
    for (int j = 0; j < 4; j++)
#pragma unroll
      for (int r = 0; r < 8; r++) {
        const int row = mbase + i * 16 + r + 8 * hi;
        const int col = nbase + j * 16 + lm;
        if (F32OUT)
          reinterpret_cast<float*>(Cout)[(size_t)row * N + col] = acc[i][j][r];
        else
          reinterpret_cast<_Float16*>(Cout)[(size_t)row * N + col] = (_Float16)acc[i][j][r];
      }
}

// ---------------------------------------------------------------- RoPE (in-place on f16 Q/K)
// Q additionally pre-scaled by 1/sqrt(HD) so attention needs no score scaling.
__global__ void rope_inplace(_Float16* __restrict__ Q, _Float16* __restrict__ Kb,
                             const int* __restrict__ pos_ids, int B, int S) {
  long long idx = (long long)blockIdx.x * blockDim.x + threadIdx.x;
  const long long total = (long long)B * S * (NHq + NKVq) * (HDq / 2);
  if (idx >= total) return;
  int d = (int)(idx % (HDq / 2)); long long t = idx / (HDq / 2);
  int head = (int)(t % (NHq + NKVq)); t /= (NHq + NKVq);
  int s = (int)(t % S); int b = (int)(t / S);
  float pos = (float)pos_ids[b * S + s];
  float inv_freq = __powf(10000.0f, -(float)(2 * d) / (float)HDq);
  float ang = pos * inv_freq;
  float cs = __cosf(ang), sn = __sinf(ang);
  const bool is_q = head < NHq;
  const float sc = is_q ? 0.08838834764831845f : 1.0f;  // 1/sqrt(128)
  _Float16* p = is_q
      ? Q  + ((size_t)b * S + s) * QH  + head * HDq
      : Kb + ((size_t)b * S + s) * KVH + (head - NHq) * HDq;
  float x1 = (float)p[d], x2 = (float)p[d + HDq / 2];
  p[d]            = (_Float16)((x1 * cs - x2 * sn) * sc);
  p[d + HDq / 2]  = (_Float16)((x2 * cs + x1 * sn) * sc);
}

// ---------------------------------------------------------------- V transpose: V[b*S+s][c] -> Vt[b*KVH+c][s]
__global__ void __launch_bounds__(256)
transpose_v(const _Float16* __restrict__ V, _Float16* __restrict__ Vt, int S) {
  __shared__ _Float16 tile[32][33];
  const int b = blockIdx.z;
  const int s0 = blockIdx.x * 32, c0 = blockIdx.y * 32;
  const int tx = threadIdx.x, ty = threadIdx.y;  // 32 x 8
#pragma unroll
  for (int i = 0; i < 32; i += 8)
    tile[ty + i][tx] = V[((size_t)b * S + s0 + ty + i) * KVH + c0 + tx];
  __syncthreads();
#pragma unroll
  for (int i = 0; i < 32; i += 8)
    Vt[((size_t)b * KVH + c0 + ty + i) * S + s0 + tx] = tile[tx][ty + i];
}

// ---------------------------------------------------------------- Flash attention (causal, GQA)
// One wave per (b, h, 16-col q tile). KV blocks of 32.
// Scores computed TRANSPOSED (S^T = K.Q^T): softmax stats reduce in-lane +
// one shfl_xor(16); P repacks in-register into the A-frag for O = P.V.
// V comes from the pre-transposed Vt buffer -> contiguous b128 B-frag loads.
__global__ void __launch_bounds__(32)
flash_attn(const _Float16* __restrict__ Q, const _Float16* __restrict__ Kb,
           const _Float16* __restrict__ Vt, _Float16* __restrict__ O, int S) {
  const int lane = threadIdx.x & 31;
  const int hi = lane >> 4;
  const int lm = lane & 15;
  const int qt = blockIdx.x, h = blockIdx.y, b = blockIdx.z;
  const int kvh = h >> 2;  // NH / NKV = 4
  const int qrow0 = qt * 16;
  const int qcol = qrow0 + lm;  // q column owned by this lane

  // Q^T as 4 B-frags: element e -> d = 32*i + 8*hi + e (e<8), 32*i + 16 + 8*hi + (e-8) (e>=8), col q = lm
  F16Frag qb[4];
  const _Float16* qp = Q + ((size_t)b * S + qrow0 + lm) * QH + h * HDq + 8 * hi;
#pragma unroll
  for (int i = 0; i < 4; i++) {
    qb[i].u[0] = *reinterpret_cast<const uint4*>(qp + 32 * i);
    qb[i].u[1] = *reinterpret_cast<const uint4*>(qp + 32 * i + 16);
  }

  v8f o[8] = {};          // O[q][d]: VGPR r -> q = r + 8*hi, lane col = d
  float m = -3e38f, l = 0.0f;

  const _Float16* kbase  = Kb + ((size_t)b * S + lm) * KVH + kvh * HDq;
  const _Float16* vtbase = Vt + ((size_t)b * KVH + kvh * HDq + lm) * S + 8 * hi;

  auto do_block = [&](int kvbase, bool masked) {
    // K rows as A-frags (2 kv tiles x 4 k-frags): issue ALL loads, then MMA
    F16Frag kf0[4], kf1[4];
    const _Float16* kp = kbase + (size_t)kvbase * KVH;
#pragma unroll
    for (int i = 0; i < 4; i++) {
      kf0[i].u[0] = *reinterpret_cast<const uint4*>(kp + 32 * i + 8 * hi);
      kf0[i].u[1] = *reinterpret_cast<const uint4*>(kp + 32 * i + 16 + 8 * hi);
      kf1[i].u[0] = *reinterpret_cast<const uint4*>(kp + (size_t)16 * KVH + 32 * i + 8 * hi);
      kf1[i].u[1] = *reinterpret_cast<const uint4*>(kp + (size_t)16 * KVH + 32 * i + 16 + 8 * hi);
    }
    __builtin_amdgcn_sched_barrier(0);
    v8f st0 = {}, st1 = {};  // S^T: kv = kvbase + (16c) + r + 8*hi, col q = qcol
#pragma unroll
    for (int i = 0; i < 4; i++) {
      st0 = wmma_f16(kf0[i].v, qb[i].v, st0);
      st1 = wmma_f16(kf1[i].v, qb[i].v, st1);
    }
    __builtin_amdgcn_sched_barrier(0);
    // issue V^T B-frag loads now; latency hides under the softmax VALU work
    F16Frag vf[8];
    const _Float16* vp = vtbase + kvbase;
#pragma unroll
    for (int j = 0; j < 8; j++) {
      vf[j].u[0] = *reinterpret_cast<const uint4*>(vp + (size_t)(16 * j) * S);
      vf[j].u[1] = *reinterpret_cast<const uint4*>(vp + (size_t)(16 * j) * S + 16);
    }
    __builtin_amdgcn_sched_barrier(0);

    // causal mask (diagonal blocks only) + in-lane max over 16 kv values
    float sv[16], vmax = -3e38f;
#pragma unroll
    for (int r = 0; r < 8; r++) {
      float x0 = st0[r];
      float x1 = st1[r];
      if (masked) {
        if (kvbase + r + 8 * hi > qcol)      x0 = -3e38f;
        if (kvbase + 16 + r + 8 * hi > qcol) x1 = -3e38f;
      }
      sv[r] = x0; sv[8 + r] = x1;
      vmax = fmaxf(vmax, fmaxf(x0, x1));
    }
    vmax = fmaxf(vmax, __shfl_xor(vmax, 16, 32));  // merge two kv half-ranges
    const float mnew = fmaxf(m, vmax);
    const float fac = __expf(m - mnew);

    // exponentiate and pack P directly as an A-frag (in-register relayout)
    F16Frag pa;
    float rsum = 0.0f;
#pragma unroll
    for (int e = 0; e < 16; e++) {
      float p = __expf(sv[e] - mnew);
      rsum += p;
      pa.v[e] = (_Float16)p;
    }
    rsum += __shfl_xor(rsum, 16, 32);
    l = l * fac + rsum;
    m = mnew;

    // broadcast per-q rescale factor into O's row layout and rescale
#pragma unroll
    for (int r = 0; r < 8; r++) {
      const float fr = __shfl(fac, r + 8 * hi, 32);
#pragma unroll
      for (int j = 0; j < 8; j++) o[j][r] *= fr;
    }

    // O += P (16x32) . V (32x128)
#pragma unroll
    for (int j = 0; j < 8; j++)
      o[j] = wmma_f16(pa.v, vf[j].v, o[j]);
  };

  const int nkvblk = (qrow0 + 47) / 32;  // blocks with kvbase <= qrow0
  const int nfull  = (qrow0 + 1) / 32;   // blocks with kvbase + 31 <= qrow0
  int kb2 = 0;
  for (; kb2 < nfull; kb2++)  do_block(kb2 * 32, false);
  for (; kb2 < nkvblk; kb2++) do_block(kb2 * 32, true);

  // normalize (broadcast l into row layout) and store f16 [B*S, NH*HD]
  _Float16* ob = O + ((size_t)b * S + qrow0) * QH + h * HDq;
#pragma unroll
  for (int r = 0; r < 8; r++) {
    const float lr = __shfl(l, r + 8 * hi, 32);
    const float inv = 1.0f / lr;
#pragma unroll
    for (int j = 0; j < 8; j++)
      ob[(size_t)(r + 8 * hi) * QH + 16 * j + lm] = (_Float16)(o[j][r] * inv);
  }
}

// ---------------------------------------------------------------- launch
extern "C" void kernel_launch(void* const* d_in, const int* in_sizes, int n_in,
                              void* d_out, int out_size, void* d_ws, size_t ws_size,
                              hipStream_t stream) {
  const int B = 2, S = 2048, H = 4096;
  const int Mrows = B * S;  // 4096

  const float* hs = (const float*)d_in[0];
  const int*  pos = (const int*)d_in[1];
  const float* wq = (const float*)d_in[2];
  const float* wk = (const float*)d_in[3];
  const float* wv = (const float*)d_in[4];
  const float* wo = (const float*)d_in[5];

  // f16 workspace layout (~209 MB)
  _Float16* X16  = (_Float16*)d_ws;
  _Float16* Wq16 = X16  + (size_t)Mrows * H;
  _Float16* Wk16 = Wq16 + (size_t)H * H;
  _Float16* Wv16 = Wk16 + (size_t)KVH * H;
  _Float16* Wo16 = Wv16 + (size_t)KVH * H;
  _Float16* Q16  = Wo16 + (size_t)H * H;
  _Float16* K16  = Q16  + (size_t)Mrows * QH;
  _Float16* V16  = K16  + (size_t)Mrows * KVH;
  _Float16* Vt16 = V16  + (size_t)Mrows * KVH;
  _Float16* AO16 = Vt16 + (size_t)Mrows * KVH;

  cast_f32_to_f16_v4<<<dim3(2048), dim3(256), 0, stream>>>(hs, X16, (long long)Mrows * H / 4);
  cast_f32_to_f16_v4<<<dim3(2048), dim3(256), 0, stream>>>(wq, Wq16, (long long)H * H / 4);
  cast_f32_to_f16_v4<<<dim3(1024), dim3(256), 0, stream>>>(wk, Wk16, (long long)KVH * H / 4);
  cast_f32_to_f16_v4<<<dim3(1024), dim3(256), 0, stream>>>(wv, Wv16, (long long)KVH * H / 4);
  cast_f32_to_f16_v4<<<dim3(2048), dim3(256), 0, stream>>>(wo, Wo16, (long long)H * H / 4);

  // projections
  gemm_xWT<0><<<dim3(H / 128,   Mrows / 128), 128, 0, stream>>>(X16, Wq16, Q16, Mrows, H,   H);
  gemm_xWT<0><<<dim3(KVH / 128, Mrows / 128), 128, 0, stream>>>(X16, Wk16, K16, Mrows, KVH, H);
  gemm_xWT<0><<<dim3(KVH / 128, Mrows / 128), 128, 0, stream>>>(X16, Wv16, V16, Mrows, KVH, H);

  // RoPE (Q pre-scaled by 1/sqrt(HD))
  {
    long long nrope = (long long)B * S * (NHq + NKVq) * (HDq / 2);
    rope_inplace<<<dim3((unsigned)((nrope + 255) / 256)), 256, 0, stream>>>(Q16, K16, pos, B, S);
  }

  // V -> V^T for contiguous PV B-frag loads
  transpose_v<<<dim3(S / 32, KVH / 32, B), dim3(32, 8), 0, stream>>>(V16, Vt16, S);

  // causal flash attention with GQA
  flash_attn<<<dim3(S / 16, NHq, B), 32, 0, stream>>>(Q16, K16, Vt16, AO16, S);

  // output projection -> fp32 d_out
  gemm_xWT<1><<<dim3(H / 128, Mrows / 128), 128, 0, stream>>>(AO16, Wo16, d_out, Mrows, H, H);
}
